// GroupAttention_12893491823210
// MI455X (gfx1250) — matrix operand order repaired and checked by reading, not compiled, not run on hardware
//
#include <hip/hip_runtime.h>
#include <cstdint>

// ---------------------------------------------------------------------------
// Types (plain ext-vectors: trivially-copyable, union-safe)
// ---------------------------------------------------------------------------
typedef __attribute__((ext_vector_type(16))) __bf16 bf16x16;
typedef __attribute__((ext_vector_type(2)))  __bf16 bf16x2;
typedef __attribute__((ext_vector_type(8)))  float  f32x8;
typedef __attribute__((ext_vector_type(4)))  float  f32x4;
typedef __attribute__((ext_vector_type(4)))  unsigned int u32x4;

union ABFrag {
  bf16x16        v;
  bf16x2         h2[8];
  u32x4          u4[2];
  unsigned short us[16];
};

// Hardware packed f32->bf16 conversion (V_CVT_PK_BF16_F32) when available.
__device__ __forceinline__ bf16x2 pk_bf16(float lo, float hi) {
#if __has_builtin(__builtin_amdgcn_cvt_pk_bf16_f32)
  return __builtin_amdgcn_cvt_pk_bf16_f32(lo, hi);
#else
  bf16x2 r;
  r[0] = (__bf16)lo;
  r[1] = (__bf16)hi;
  return r;
#endif
}

__device__ __forceinline__ unsigned short f2bf(float f) {
  __bf16 b = (__bf16)f;                 // RNE, lowers to hw convert
  unsigned short u; __builtin_memcpy(&u, &b, 2);
  return u;
}
__device__ __forceinline__ float bf2f(unsigned short u) {
  unsigned int x = ((unsigned int)u) << 16;
  float f; __builtin_memcpy(&f, &x, 4);
  return f;
}

__device__ __forceinline__ f32x8 wmma_bf16(const ABFrag& a, const ABFrag& b, f32x8 c) {
  // D = A(16x32 bf16) * B(32x16 bf16) + C(16x16 f32)
  return __builtin_amdgcn_wmma_f32_16x16x32_bf16(
      /*neg_a=*/false, a.v, /*neg_b=*/false, b.v,
      /*c_mod=*/(short)0, c, /*reuse_a=*/false, /*reuse_b=*/false);
}

// Async global->LDS copy, 16B per lane (CDNA5 GLOBAL_LOAD_ASYNC_TO_LDS_B128,
// tracked by ASYNCcnt). lds_off = per-lane LDS byte address, g = global address.
__device__ __forceinline__ void async_cp16(unsigned int lds_off, const void* g) {
  asm volatile("global_load_async_to_lds_b128 %0, %1, off"
               :: "v"(lds_off), "v"((unsigned long long)(uintptr_t)g)
               : "memory");
}
__device__ __forceinline__ void wait_async0() {
  asm volatile("s_wait_asynccnt 0x0" ::: "memory");
}

// Build a 16x32 bf16 A fragment from an fp32 row (K-contiguous).
// ISA layout: lane<16 -> elems 0..7 = K kb*32+0..7, elems 8..15 = K kb*32+16..23
//             lane>=16 -> same +8.
__device__ __forceinline__ void load_a_f32(ABFrag& f, const float* row, int kb, int hi) {
  const int k0 = kb * 32 + hi * 8;
  f32x4 a0 = *(const f32x4*)(row + k0);
  f32x4 a1 = *(const f32x4*)(row + k0 + 4);
  f32x4 a2 = *(const f32x4*)(row + k0 + 16);
  f32x4 a3 = *(const f32x4*)(row + k0 + 20);
  f.h2[0] = pk_bf16(a0.x, a0.y);
  f.h2[1] = pk_bf16(a0.z, a0.w);
  f.h2[2] = pk_bf16(a1.x, a1.y);
  f.h2[3] = pk_bf16(a1.z, a1.w);
  f.h2[4] = pk_bf16(a2.x, a2.y);
  f.h2[5] = pk_bf16(a2.z, a2.w);
  f.h2[6] = pk_bf16(a3.x, a3.y);
  f.h2[7] = pk_bf16(a3.z, a3.w);
}

// ---------------------------------------------------------------------------
// Kernel 0: convert weights fp32 -> bf16 into workspace
// ---------------------------------------------------------------------------
__global__ __launch_bounds__(256) void cvt_weights_kernel(
    const float* __restrict__ wqkv_f, const float* __restrict__ wproj_f,
    unsigned short* __restrict__ wqkv_b, unsigned short* __restrict__ wproj_b,
    int n_qkv, int n_proj) {
  int i = blockIdx.x * 256 + threadIdx.x;   // element-pair index
  int j = i * 2;
  if (j + 1 < n_qkv) {
    bf16x2 p = pk_bf16(wqkv_f[j], wqkv_f[j + 1]);
    unsigned int u; __builtin_memcpy(&u, &p, 4);
    *(unsigned int*)(wqkv_b + j) = u;
  }
  if (j + 1 < n_proj) {
    bf16x2 p = pk_bf16(wproj_f[j], wproj_f[j + 1]);
    unsigned int u; __builtin_memcpy(&u, &p, 4);
    *(unsigned int*)(wproj_b + j) = u;
  }
}

// ---------------------------------------------------------------------------
// Kernel 1: fused grouped-QKV (WMMA) + tiny 4x4 attention.
// One workgroup = 16 tokens, 256 threads = 8 waves. LDS = 64 KB static.
//   Phase 1: compute Q,K (e in [0,512)) for all 4 groups -> bf16 LDS qk[16][2048]
//   Phase 2: per-(token,head,p) softmax over the 4 channel groups -> attn LDS (4KB,
//            written over the then-dead qk region)
//   Phase 3: compute V tiles with WMMA, combine with attn, store bf16 att rows.
// ---------------------------------------------------------------------------
__global__ __launch_bounds__(256) void qkv_attn_kernel(
    const float* __restrict__ x,              // (tokens, 1024) fp32
    const unsigned short* __restrict__ wqkv,  // (768, 256) bf16, K-contiguous
    unsigned short* __restrict__ att)         // (tokens, 1024) bf16 out
{
  __shared__ __align__(16) char smem[65536];
  unsigned short (*qk)[2048] = (unsigned short (*)[2048])smem;  // [tok][g*512 + s*256 + h*64 + d]
  float* attn = (float*)smem;                                   // [((tok*4+h)*4+p)*4+q], after qk dead

  const int tid  = threadIdx.x;
  const int lane = tid & 31;
  const int wv   = tid >> 5;
  const int hi   = lane >> 4;      // half-wave select
  const int l15  = lane & 15;
  const long tok0 = (long)blockIdx.x * 16;
  const float* xbase = x + tok0 * 1024;

  // ---------------- Phase 1: Q,K tiles ----------------
  {
    const int g    = wv & 3;   // channel group
    const int half = wv >> 2;  // which 16 of the 32 e-tiles
    ABFrag af[8];
    const float* xrow = xbase + (long)l15 * 1024 + g * 256;
#pragma unroll
    for (int kb = 0; kb < 8; ++kb) load_a_f32(af[kb], xrow, kb, hi);

    for (int nt = 0; nt < 16; ++nt) {
      const int e0 = (half * 16 + nt) * 16;                  // e in [0,512): Q then K
      const unsigned short* wrow = wqkv + (long)(e0 + l15) * 256;
      f32x8 acc = {0.f, 0.f, 0.f, 0.f, 0.f, 0.f, 0.f, 0.f};
#pragma unroll
      for (int kb = 0; kb < 8; ++kb) {
        ABFrag bfr;
        const int k = kb * 32 + hi * 8;
        bfr.u4[0] = *(const u32x4*)(wrow + k);
        bfr.u4[1] = *(const u32x4*)(wrow + k + 16);
        acc = wmma_bf16(af[kb], bfr, acc);
      }
      const int col = g * 512 + e0 + l15;
#pragma unroll
      for (int r = 0; r < 8; ++r) qk[r + hi * 8][col] = f2bf(acc[r]);
    }
  }
  __syncthreads();

  // ---------------- Phase 2: softmax over the 4 groups ----------------
  {
    const int tok = tid >> 4;
    const int h   = (tid >> 2) & 3;
    const int p   = tid & 3;
    const unsigned short* qrow = &qk[tok][p * 512 + h * 64];
    float s[4];
    for (int qi = 0; qi < 4; ++qi) {
      const unsigned short* krow = &qk[tok][qi * 512 + 256 + h * 64];
      float acc = 0.f;
      for (int d = 0; d < 64; ++d) acc += bf2f(qrow[d]) * bf2f(krow[d]);
      s[qi] = acc * 0.125f;  // hd^-0.5, hd=64
    }
    float m  = fmaxf(fmaxf(s[0], s[1]), fmaxf(s[2], s[3]));
    float e0 = __expf(s[0] - m), e1 = __expf(s[1] - m);
    float e2 = __expf(s[2] - m), e3 = __expf(s[3] - m);
    float inv = 1.f / (e0 + e1 + e2 + e3);
    __syncthreads();   // qk region now dead; safe to overwrite with attn
    float* ap = attn + ((tok * 4 + h) * 4 + p) * 4;
    ap[0] = e0 * inv; ap[1] = e1 * inv; ap[2] = e2 * inv; ap[3] = e3 * inv;
  }
  __syncthreads();

  // ---------------- Phase 3: V tiles + weighted combine ----------------
  for (int i = 0; i < 2; ++i) {
    const int pair = wv * 2 + i;   // 16 (h, dt) pairs over 8 waves
    const int h  = pair >> 2;
    const int dt = pair & 3;
    // B frags for V columns (e = 512 + h*64 + dt*16 + n) — identical for every group
    ABFrag bv[8];
    const unsigned short* wrow = wqkv + (long)(512 + h * 64 + dt * 16 + l15) * 256;
#pragma unroll
    for (int kb = 0; kb < 8; ++kb) {
      const int k = kb * 32 + hi * 8;
      bv[kb].u4[0] = *(const u32x4*)(wrow + k);
      bv[kb].u4[1] = *(const u32x4*)(wrow + k + 16);
    }
    float outp[4][8];
#pragma unroll
    for (int pp = 0; pp < 4; ++pp)
#pragma unroll
      for (int r = 0; r < 8; ++r) outp[pp][r] = 0.f;

#pragma unroll
    for (int g = 0; g < 4; ++g) {
      const float* xrow = xbase + (long)l15 * 1024 + g * 256;
      f32x8 acc = {0.f, 0.f, 0.f, 0.f, 0.f, 0.f, 0.f, 0.f};
#pragma unroll
      for (int kb = 0; kb < 8; ++kb) {
        ABFrag a;
        load_a_f32(a, xrow, kb, hi);
        acc = wmma_bf16(a, bv[kb], acc);
      }
#pragma unroll
      for (int pp = 0; pp < 4; ++pp)
#pragma unroll
        for (int r = 0; r < 8; ++r) {
          const int tokr = r + hi * 8;
          outp[pp][r] += attn[((tokr * 4 + h) * 4 + pp) * 4 + g] * acc[r];
        }
    }
#pragma unroll
    for (int pp = 0; pp < 4; ++pp)
#pragma unroll
      for (int r = 0; r < 8; ++r) {
        const int tokr = r + hi * 8;
        att[(tok0 + tokr) * 1024 + pp * 256 + h * 64 + dt * 16 + l15] = f2bf(outp[pp][r]);
      }
  }
}

// ---------------------------------------------------------------------------
// Kernel 2: out = att(bf16) @ w_proj^T(bf16) + b  (M=tokens, N=K=1024)
// Workgroup tile 64x128, 8 waves, wave tile 32x32 (2x2 WMMA accumulators).
// A/B tiles staged in LDS via async global->LDS copies (ASYNCcnt), double
// buffered: prefetch K-step kb+1 while WMMAs consume kb from LDS.
// LDS: 2 * (A 64x32 bf16 [4KB] + B 128x32 bf16 [8KB]) = 24 KB.
// ---------------------------------------------------------------------------
__global__ __launch_bounds__(256) void proj_kernel(
    const unsigned short* __restrict__ att,    // (tokens, 1024) bf16
    const unsigned short* __restrict__ wproj,  // (1024, 1024) bf16, K-contiguous
    const float* __restrict__ bias,            // (1024,)
    float* __restrict__ out)                   // (tokens, 1024) fp32
{
  __shared__ __align__(16) char lds[24576];
  // buffer b: A at lds + b*4096 (row stride 64B), B at lds + 8192 + b*8192

  const int tid  = threadIdx.x;
  const int lane = tid & 31;
  const int wv   = tid >> 5;
  const int hi   = lane >> 4;
  const int l15  = lane & 15;
  const int m0   = blockIdx.y * 64;
  const int n0   = blockIdx.x * 128;

  const unsigned short* att_blk = att + (long)m0 * 1024;
  const unsigned short* wp_blk  = wproj + (long)n0 * 1024;

  // Per-thread async copy chunk assignment (16B chunks, 32 bf16 per row):
  //   A: 256 chunks -> 1/thread;  B: 512 chunks -> 2/thread.
  const int ca  = tid;              // A: row = ca>>2 (0..63), sub = ca&3
  const int cb0 = tid;              // B: rows 0..63
  const int cb1 = tid + 256;        // B: rows 64..127
  const unsigned int ldsA_off =
      (unsigned int)(uintptr_t)(lds + (ca >> 2) * 64 + (ca & 3) * 16);
  const unsigned int ldsB0_off =
      (unsigned int)(uintptr_t)(lds + 8192 + (cb0 >> 2) * 64 + (cb0 & 3) * 16);
  const unsigned int ldsB1_off =
      (unsigned int)(uintptr_t)(lds + 8192 + (cb1 >> 2) * 64 + (cb1 & 3) * 16);
  const unsigned short* gA  = att_blk + (long)(ca >> 2) * 1024 + (ca & 3) * 8;
  const unsigned short* gB0 = wp_blk + (long)(cb0 >> 2) * 1024 + (cb0 & 3) * 8;
  const unsigned short* gB1 = wp_blk + (long)(cb1 >> 2) * 1024 + (cb1 & 3) * 8;

#define FILL_TILE(buf, kb)                                        \
  do {                                                            \
    async_cp16(ldsA_off + (buf) * 4096, gA + (kb) * 32);          \
    async_cp16(ldsB0_off + (buf) * 8192, gB0 + (kb) * 32);        \
    async_cp16(ldsB1_off + (buf) * 8192, gB1 + (kb) * 32);        \
  } while (0)

  f32x8 acc00 = {0.f,0.f,0.f,0.f,0.f,0.f,0.f,0.f};
  f32x8 acc01 = acc00, acc10 = acc00, acc11 = acc00;

  // LDS fragment base addresses for this wave
  const char* pa = lds + ((wv >> 2) * 32 + l15) * 64 + hi * 16;           // A rows
  const char* pb = lds + 8192 + ((wv & 3) * 32 + l15) * 64 + hi * 16;     // B rows

  FILL_TILE(0, 0);
  wait_async0();
  __syncthreads();

#pragma unroll 2
  for (int kb = 0; kb < 32; ++kb) {
    const int buf = kb & 1;
    if (kb + 1 < 32) FILL_TILE(buf ^ 1, kb + 1);   // async prefetch next K-step

    ABFrag a0, a1, b0, b1;
    const char* pab = pa + buf * 4096;
    const char* pbb = pb + buf * 8192;
    a0.u4[0] = *(const u32x4*)(pab);
    a0.u4[1] = *(const u32x4*)(pab + 32);
    a1.u4[0] = *(const u32x4*)(pab + 16 * 64);
    a1.u4[1] = *(const u32x4*)(pab + 16 * 64 + 32);
    b0.u4[0] = *(const u32x4*)(pbb);
    b0.u4[1] = *(const u32x4*)(pbb + 32);
    b1.u4[0] = *(const u32x4*)(pbb + 16 * 64);
    b1.u4[1] = *(const u32x4*)(pbb + 16 * 64 + 32);

    acc00 = wmma_bf16(a0, b0, acc00);
    acc01 = wmma_bf16(a0, b1, acc01);
    acc10 = wmma_bf16(a1, b0, acc10);
    acc11 = wmma_bf16(a1, b1, acc11);

    wait_async0();       // own wave's async copies for kb+1 done
    __syncthreads();     // all waves done reading buf / writing buf^1
  }
#undef FILL_TILE

  const int mw = m0 + (wv >> 2) * 32;
  const int nw = n0 + (wv & 3) * 32;
  const float bias0 = bias[nw + l15];
  const float bias1 = bias[nw + 16 + l15];
#pragma unroll
  for (int r = 0; r < 8; ++r) {
    const int row0 = mw + r + hi * 8;
    const int row1 = row0 + 16;
    out[(long)row0 * 1024 + nw + l15]      = acc00[r] + bias0;
    out[(long)row0 * 1024 + nw + 16 + l15] = acc01[r] + bias1;
    out[(long)row1 * 1024 + nw + l15]      = acc10[r] + bias0;
    out[(long)row1 * 1024 + nw + 16 + l15] = acc11[r] + bias1;
  }
}

// ---------------------------------------------------------------------------
// Host launcher
// ---------------------------------------------------------------------------
extern "C" void kernel_launch(void* const* d_in, const int* in_sizes, int n_in,
                              void* d_out, int out_size, void* d_ws, size_t ws_size,
                              hipStream_t stream) {
  const float* x      = (const float*)d_in[0];   // (B,N,C)
  const float* wqkv_f = (const float*)d_in[1];   // (768,256)
  const float* wproj_f= (const float*)d_in[2];   // (1024,1024)
  const float* bproj  = (const float*)d_in[3];   // (1024,)

  const int C      = 1024;
  const int tokens = in_sizes[0] / C;            // B*N = 32768
  const int n_qkv  = in_sizes[1];                // 196608
  const int n_proj = in_sizes[2];                // 1048576

  char* ws = (char*)d_ws;
  size_t off_qkv  = 0;
  size_t off_proj = ((size_t)n_qkv * 2 + 255) & ~(size_t)255;
  size_t off_att  = (off_proj + (size_t)n_proj * 2 + 255) & ~(size_t)255;
  unsigned short* wqkv_b  = (unsigned short*)(ws + off_qkv);
  unsigned short* wproj_b = (unsigned short*)(ws + off_proj);
  unsigned short* att     = (unsigned short*)(ws + off_att);
  float* out = (float*)d_out;

  // 0) weights -> bf16 (paired converts)
  int n_cvt   = (n_proj > n_qkv) ? n_proj : n_qkv;
  int n_pairs = (n_cvt + 1) / 2;
  cvt_weights_kernel<<<(n_pairs + 255) / 256, 256, 0, stream>>>(
      wqkv_f, wproj_f, wqkv_b, wproj_b, n_qkv, n_proj);

  // 1) fused grouped-QKV + attention -> att (bf16)
  qkv_attn_kernel<<<tokens / 16, 256, 0, stream>>>(x, wqkv_b, att);

  // 2) output projection GEMM (+bias) -> out (fp32)
  proj_kernel<<<dim3(C / 128, tokens / 64), 256, 0, stream>>>(att, wproj_b, bproj, out);
}
